// SpatialAttention_18889266168048
// MI455X (gfx1250) — matrix operand order repaired and checked by reading, not compiled, not run on hardware
//
#include <hip/hip_runtime.h>
#include <cstddef>
#include <cstdint>

#define B_  8
#define C_  256
#define CO_ 128
#define N_  4096   // H*W = 64*64

typedef __attribute__((ext_vector_type(16))) _Float16 v16h;
typedef __attribute__((ext_vector_type(8)))  _Float16 v8h;
typedef __attribute__((ext_vector_type(8)))  float    v8f;

#define LDH 40    // padded LDS row stride for 32-half tiles (80B rows, 16B aligned)
#define LDE 136   // padded LDS row stride for 128-half tiles (272B = 68 dwords ≡ 4 mod 64 banks)

static __device__ __forceinline__ v16h cat16(v8h lo, v8h hi) {
  return __builtin_shufflevector(lo, hi, 0,1,2,3,4,5,6,7,8,9,10,11,12,13,14,15);
}

// A-matrix 16x32 f16 fragment (ISA 7.12.2): lanes 0-15: K in {0..7,16..23};
// lanes 16-31: K in {8..15,24..31}. p points at row start (K offset 0).
static __device__ __forceinline__ v16h fragA(const _Float16* p, int lane) {
  p += (lane >> 4) << 3;
  return cat16(*(const v8h*)p, *(const v8h*)(p + 16));
}
// B-matrix 32x16 f16 fragment: lanes 0-15 K in {0..7,16..23}, lanes 16-31 K in {8..15,24..31},
// from the [Nrow][K] staged tile. p points at N-row start.
static __device__ __forceinline__ v16h fragB(const _Float16* p, int lane) {
  p += (lane >> 4) << 3;
  return cat16(*(const v8h*)p, *(const v8h*)(p + 16));
}

#define WMMA_F16(a,b,c) \
  __builtin_amdgcn_wmma_f32_16x16x32_f16(false,(a),false,(b),(short)0,(c),false,false)

// CDNA5 async memory->LDS copy (ASYNCcnt). offset: is added to BOTH the global
// and the LDS address (ISA 08, §4.4), so one address pair serves a whole row run.
#define ASYNC_B128(lds, ga, OFF)                                              \
  asm volatile("global_load_async_to_lds_b128 %0, %1, off offset:" #OFF       \
               :: "v"(lds), "v"(ga) : "memory")
#define WAIT_ASYNC0() asm volatile("s_wait_asynccnt 0x0" ::: "memory")

static __device__ __forceinline__ v8h cvt8(const float4 a, const float4 b) {
  v8h h;
  h[0]=(_Float16)a.x; h[1]=(_Float16)a.y; h[2]=(_Float16)a.z; h[3]=(_Float16)a.w;
  h[4]=(_Float16)b.x; h[5]=(_Float16)b.y; h[6]=(_Float16)b.z; h[7]=(_Float16)b.w;
  return h;
}

// ---------------------------------------------------------------------------
// 1) Projection: Y[b][n][o] = sum_c W[o][c] * X[b][c][n] + bias[o]   (f16 out)
//    128n x 128o block tile, K over C in 32s, double-buffered LDS.
// ---------------------------------------------------------------------------
__global__ __launch_bounds__(256)
void proj_kernel(const float* __restrict__ X,   // [B, C, N]
                 const float* __restrict__ Wt,  // [CO, C]
                 const float* __restrict__ bias,// [CO]
                 _Float16* __restrict__ Y)      // [B, N, CO]
{
  __shared__ _Float16 Al[2][128 * LDH];   // X^T tile: [n 128][c 32]
  __shared__ _Float16 Bl[2][128 * LDH];   // W tile:   [o 128][c 32]
  const int b    = blockIdx.y;
  const int nb   = blockIdx.x * 128;
  const int t    = threadIdx.x;
  const int lane = t & 31;
  const int w    = t >> 5;
  const int wr   = w & 3;
  const int wc   = w >> 2;

  const float* Xb = X + (size_t)b * C_ * N_;

  auto stage = [&](int buf, int ck) {
    { // A: read X[c][n] 32x128 coalesced over n, transpose+cvt into Al[n][c]
      const int c  = t >> 3;
      const int nn = (t & 7) * 16;
      const float* src = Xb + (size_t)(ck + c) * N_ + nb + nn;
      #pragma unroll
      for (int i = 0; i < 16; ++i) Al[buf][(nn + i) * LDH + c] = (_Float16)src[i];
    }
    { // B: W[o][ck..ck+32) contiguous rows, float4 loads -> packed b128 LDS stores
      const int o  = t >> 1;
      const int cs = (t & 1) * 16;
      const float4* src = (const float4*)(Wt + (size_t)o * C_ + ck + cs);
      *(v8h*)&Bl[buf][o * LDH + cs]     = cvt8(src[0], src[1]);
      *(v8h*)&Bl[buf][o * LDH + cs + 8] = cvt8(src[2], src[3]);
    }
  };

  const v8f vz = {0.f,0.f,0.f,0.f,0.f,0.f,0.f,0.f};
  v8f acc[2][4];
  #pragma unroll
  for (int i = 0; i < 2; ++i)
    #pragma unroll
    for (int j = 0; j < 4; ++j) acc[i][j] = vz;

  stage(0, 0);
  for (int s = 0; s < 8; ++s) {
    __syncthreads();
    if (s < 7) stage((s + 1) & 1, (s + 1) * 32);
    const _Float16* A = Al[s & 1];
    const _Float16* Bt = Bl[s & 1];
    v16h af[2], bf[4];
    #pragma unroll
    for (int i = 0; i < 2; ++i) af[i] = fragA(A + (wr*32 + i*16 + (lane & 15)) * LDH, lane);
    #pragma unroll
    for (int j = 0; j < 4; ++j) bf[j] = fragB(Bt + (wc*64 + j*16 + (lane & 15)) * LDH, lane);
    #pragma unroll
    for (int i = 0; i < 2; ++i)
      #pragma unroll
      for (int j = 0; j < 4; ++j) acc[i][j] = WMMA_F16(af[i], bf[j], acc[i][j]);
  }

  _Float16* Yb = Y + (size_t)b * N_ * CO_;
  #pragma unroll
  for (int j = 0; j < 4; ++j) {
    const int o  = wc * 64 + j * 16 + (lane & 15);
    const float bv = bias[o];
    #pragma unroll
    for (int i = 0; i < 2; ++i) {
      #pragma unroll
      for (int r = 0; r < 8; ++r) {
        const int n = nb + wr * 32 + i * 16 + ((lane >> 4) << 3) + r;
        Yb[(size_t)n * CO_ + o] = (_Float16)(acc[i][j][r] + bv);
      }
    }
  }
}

// ---------------------------------------------------------------------------
// 2) Energy: E[b][n][m] = sum_o Q[b][n][o] * K[b][m][o]   (f32 -> d_out att)
//    Full K-depth (CO=128) staged once via async global->LDS b128 copies:
//    one s_wait_asynccnt + one barrier per block, then 32 WMMAs/wave.
// ---------------------------------------------------------------------------
__global__ __launch_bounds__(256)
void energy_kernel(const _Float16* __restrict__ Q,  // [B, N, CO]
                   const _Float16* __restrict__ Kt, // [B, M, CO]
                   float* __restrict__ E)           // [B, N, N]
{
  __shared__ _Float16 Al[128 * LDE];   // Q tile [n 128][o 128]
  __shared__ _Float16 Bl[128 * LDE];   // K tile [m 128][o 128]
  const int b    = blockIdx.z;
  const int nb   = blockIdx.y * 128;
  const int mb   = blockIdx.x * 128;
  const int t    = threadIdx.x;
  const int lane = t & 31;
  const int w    = t >> 5;
  const int wr   = w & 3;
  const int wc   = w >> 2;

  // each thread copies 128B of one row of each tile: 8 async b128 ops per tile
  const int r  = t >> 1;            // row 0..127
  const int hh = (t & 1) * 64;      // half-offset within row: 0 or 64 (=128B)

  const _Float16* Qb = Q  + ((size_t)b * N_ + nb) * CO_;
  const _Float16* Kb = Kt + ((size_t)b * N_ + mb) * CO_;

  const uint64_t gA = (uint64_t)(uintptr_t)(Qb + (size_t)r * CO_ + hh);
  const uint64_t gB = (uint64_t)(uintptr_t)(Kb + (size_t)r * CO_ + hh);
  // low 32 bits of a flat LDS pointer are the LDS byte address (ISA 10.2 aperture map)
  const uint32_t lA = (uint32_t)(uintptr_t)(Al + r * LDE + hh);
  const uint32_t lB = (uint32_t)(uintptr_t)(Bl + r * LDE + hh);

  ASYNC_B128(lA, gA, 0);   ASYNC_B128(lA, gA, 16);
  ASYNC_B128(lA, gA, 32);  ASYNC_B128(lA, gA, 48);
  ASYNC_B128(lA, gA, 64);  ASYNC_B128(lA, gA, 80);
  ASYNC_B128(lA, gA, 96);  ASYNC_B128(lA, gA, 112);
  ASYNC_B128(lB, gB, 0);   ASYNC_B128(lB, gB, 16);
  ASYNC_B128(lB, gB, 32);  ASYNC_B128(lB, gB, 48);
  ASYNC_B128(lB, gB, 64);  ASYNC_B128(lB, gB, 80);
  ASYNC_B128(lB, gB, 96);  ASYNC_B128(lB, gB, 112);

  const v8f vz = {0.f,0.f,0.f,0.f,0.f,0.f,0.f,0.f};
  v8f acc[2][4];
  #pragma unroll
  for (int i = 0; i < 2; ++i)
    #pragma unroll
    for (int j = 0; j < 4; ++j) acc[i][j] = vz;

  WAIT_ASYNC0();        // this wave's async copies landed in LDS
  __syncthreads();      // all waves' copies visible

  #pragma unroll
  for (int ko = 0; ko < CO_; ko += 32) {
    v16h af[2], bf[4];
    #pragma unroll
    for (int i = 0; i < 2; ++i)
      af[i] = fragA(Al + (wr*32 + i*16 + (lane & 15)) * LDE + ko, lane);
    #pragma unroll
    for (int j = 0; j < 4; ++j)
      bf[j] = fragB(Bl + (wc*64 + j*16 + (lane & 15)) * LDE + ko, lane);
    #pragma unroll
    for (int i = 0; i < 2; ++i)
      #pragma unroll
      for (int j = 0; j < 4; ++j) acc[i][j] = WMMA_F16(af[i], bf[j], acc[i][j]);
  }

  float* Eb = E + (size_t)b * N_ * N_;
  #pragma unroll
  for (int i = 0; i < 2; ++i) {
    #pragma unroll
    for (int r2 = 0; r2 < 8; ++r2) {
      const int n = nb + wr * 32 + i * 16 + ((lane >> 4) << 3) + r2;
      #pragma unroll
      for (int j = 0; j < 4; ++j) {
        const int m = mb + wc * 64 + j * 16 + (lane & 15);
        Eb[(size_t)n * N_ + m] = acc[i][j][r2];
      }
    }
  }
}

// ---------------------------------------------------------------------------
// 3) Softmax: one 256-thread block per row of 4096 (in place, f32)
// ---------------------------------------------------------------------------
__global__ __launch_bounds__(256)
void softmax_kernel(float* __restrict__ E)   // [B*N, N]
{
  __shared__ float red[8];
  float* p = E + (size_t)blockIdx.x * N_;
  const int t = threadIdx.x;

  float v[16];
  float mx = -3.402823466e+38f;
  #pragma unroll
  for (int i = 0; i < 16; ++i) { v[i] = p[t + i * 256]; mx = fmaxf(mx, v[i]); }

  #pragma unroll
  for (int o = 16; o > 0; o >>= 1) mx = fmaxf(mx, __shfl_xor(mx, o, 32));
  if ((t & 31) == 0) red[t >> 5] = mx;
  __syncthreads();
  float m = red[0];
  #pragma unroll
  for (int i = 1; i < 8; ++i) m = fmaxf(m, red[i]);
  __syncthreads();

  float s = 0.f;
  #pragma unroll
  for (int i = 0; i < 16; ++i) { v[i] = __expf(v[i] - m); s += v[i]; }
  #pragma unroll
  for (int o = 16; o > 0; o >>= 1) s += __shfl_xor(s, o, 32);
  if ((t & 31) == 0) red[t >> 5] = s;
  __syncthreads();
  float sum = 0.f;
  #pragma unroll
  for (int i = 0; i < 8; ++i) sum += red[i];
  const float inv = 1.f / sum;

  #pragma unroll
  for (int i = 0; i < 16; ++i) p[t + i * 256] = v[i] * inv;
}

// ---------------------------------------------------------------------------
// 4) Output: Out[b][c][n] = sum_m V[b][c][m] * Att[b][n][m]
//    Double-buffered LDS, f32->f16 cvt during staging, one barrier per K-step.
// ---------------------------------------------------------------------------
__global__ __launch_bounds__(256)
void out_kernel(const float* __restrict__ Att, // [B, N, N]
                const float* __restrict__ V,   // [B, C, N]
                float* __restrict__ Out)       // [B, C, N]
{
  __shared__ _Float16 Al[2][128 * LDH];   // V tile   [c 128][m 32]
  __shared__ _Float16 Bl[2][128 * LDH];   // Att tile [n 128][m 32]
  const int b    = blockIdx.z;
  const int cb   = blockIdx.y * 128;
  const int nb   = blockIdx.x * 128;
  const int t    = threadIdx.x;
  const int lane = t & 31;
  const int w    = t >> 5;
  const int wr   = w & 3;
  const int wc   = w >> 2;

  const float* Vb = V   + (size_t)b * C_ * N_;
  const float* Ab = Att + (size_t)b * N_ * N_;

  const int rr = t >> 1;
  const int hh = (t & 1) * 16;

  auto stage = [&](int buf, int ko) {
    {
      const float4* src = (const float4*)(Vb + (size_t)(cb + rr) * N_ + ko + hh);
      *(v8h*)&Al[buf][rr * LDH + hh]     = cvt8(src[0], src[1]);
      *(v8h*)&Al[buf][rr * LDH + hh + 8] = cvt8(src[2], src[3]);
    }
    {
      const float4* src = (const float4*)(Ab + (size_t)(nb + rr) * N_ + ko + hh);
      *(v8h*)&Bl[buf][rr * LDH + hh]     = cvt8(src[0], src[1]);
      *(v8h*)&Bl[buf][rr * LDH + hh + 8] = cvt8(src[2], src[3]);
    }
  };

  const v8f vz = {0.f,0.f,0.f,0.f,0.f,0.f,0.f,0.f};
  v8f acc[2][4];
  #pragma unroll
  for (int i = 0; i < 2; ++i)
    #pragma unroll
    for (int j = 0; j < 4; ++j) acc[i][j] = vz;

  stage(0, 0);
  for (int s = 0; s < 128; ++s) {
    __syncthreads();
    if (s < 127) stage((s + 1) & 1, (s + 1) * 32);
    const _Float16* A = Al[s & 1];
    const _Float16* Bt = Bl[s & 1];
    v16h af[2], bf[4];
    #pragma unroll
    for (int i = 0; i < 2; ++i) af[i] = fragA(A + (wr*32 + i*16 + (lane & 15)) * LDH, lane);
    #pragma unroll
    for (int j = 0; j < 4; ++j) bf[j] = fragB(Bt + (wc*64 + j*16 + (lane & 15)) * LDH, lane);
    #pragma unroll
    for (int i = 0; i < 2; ++i)
      #pragma unroll
      for (int j = 0; j < 4; ++j) acc[i][j] = WMMA_F16(af[i], bf[j], acc[i][j]);
  }

  // coalesced stores: fixed c per VGPR row, n contiguous across 16 lanes
  #pragma unroll
  for (int i = 0; i < 2; ++i) {
    #pragma unroll
    for (int r = 0; r < 8; ++r) {
      const int c = cb + wr * 32 + i * 16 + ((lane >> 4) << 3) + r;
      #pragma unroll
      for (int j = 0; j < 4; ++j) {
        const int n = nb + wc * 64 + j * 16 + (lane & 15);
        Out[((size_t)b * C_ + c) * N_ + n] = acc[i][j][r];
      }
    }
  }
}

// ---------------------------------------------------------------------------
extern "C" void kernel_launch(void* const* d_in, const int* in_sizes, int n_in,
                              void* d_out, int out_size, void* d_ws, size_t ws_size,
                              hipStream_t stream)
{
  (void)in_sizes; (void)n_in; (void)out_size; (void)ws_size;
  const float* p  = (const float*)d_in[0];   // [B, C, H, W]
  const float* bm = (const float*)d_in[1];   // [B, C, H, W]
  const float* Wq = (const float*)d_in[2];   // [CO, C]
  const float* bq = (const float*)d_in[3];   // [CO]
  const float* Wk = (const float*)d_in[4];   // [CO, C]
  const float* bk = (const float*)d_in[5];   // [CO]

  float* out = (float*)d_out;                       // out  [B, C, N]
  float* att = out + (size_t)B_ * C_ * N_;          // attn [B, N, N]

  _Float16* qh = (_Float16*)d_ws;                   // [B, N, CO]
  _Float16* kh = qh + (size_t)B_ * N_ * CO_;        // [B, M, CO]  (16.8 MB total)

  const dim3 blk(256);
  proj_kernel   <<<dim3(N_/128, B_),          blk, 0, stream>>>(p,  Wq, bq, qh);
  proj_kernel   <<<dim3(N_/128, B_),          blk, 0, stream>>>(bm, Wk, bk, kh);
  energy_kernel <<<dim3(N_/128, N_/128, B_),  blk, 0, stream>>>(qh, kh, att);
  softmax_kernel<<<dim3(B_ * N_),             blk, 0, stream>>>(att);
  out_kernel    <<<dim3(N_/128, C_/128, B_),  blk, 0, stream>>>(att, bm, out);
}
